// RNNLayers_22067541967800
// MI455X (gfx1250) — compile-verified, matrix operand order checked by reading, not run on hardware
//
#include <hip/hip_runtime.h>

// Problem constants (match reference)
#define LAYERS 4
#define NB     32      // batch
#define SEQ    2048
#define VD     512     // input dim
#define DD     512     // hidden dim
#define GPL    8       // workgroups per layer
#define ECOLS  (DD / GPL)   // 64 output columns per workgroup

typedef __attribute__((ext_vector_type(16))) __bf16 v16bf;
typedef __attribute__((ext_vector_type(8)))  float  v8f;

__device__ __forceinline__ unsigned short f2bf(float f) {
    union { __bf16 b; unsigned short u; } cv;
    cv.b = (__bf16)f;                     // native f32->bf16 cvt (RNE)
    return cv.u;
}

__device__ __forceinline__ unsigned int f2bf_pk(float lo, float hi) {
    return (unsigned int)f2bf(lo) | ((unsigned int)f2bf(hi) << 16);
}

// Load a 16x32 bf16 fragment (A or B operand) from a row-major [rows][512]
// bf16 LDS array, following the CDNA5 16-bit operand striping:
// lanes 0-15 hold K 0..7 (v0-3) and 16..23 (v4-7); lanes 16-31 hold K 8..15, 24..31.
__device__ __forceinline__ v16bf load_frag(const unsigned short* base, int row, int kt, int lane) {
    int kb = (lane < 16) ? 0 : 8;
    const unsigned short* p = base + row * 512 + kt * 32 + kb;
    union { v16bf v; uint4 q[2]; } u;
    u.q[0] = *(const uint4*)(p);
    u.q[1] = *(const uint4*)(p + 16);
    return u.v;
}

__device__ __forceinline__ v8f wmma_bf16(v16bf a, v16bf b, v8f c) {
    return __builtin_amdgcn_wmma_f32_16x16x32_bf16(
        /*neg_a=*/false, a, /*neg_b=*/false, b,
        /*c_mod=*/(short)0, c, /*reuse_a=*/false, /*reuse_b=*/false);
}

// Stage x[:, s, :] (f32 global) into bf16 LDS tile [NB][VD].
__device__ __forceinline__ void stage_x(const float* __restrict__ x,
                                        unsigned short* __restrict__ x_s,
                                        int s, int tid) {
    int n  = tid >> 3;
    int vb = (tid & 7) << 6;
    const float* src = x + ((size_t)n * SEQ + s) * VD + vb;
    unsigned short* dst = x_s + n * VD + vb;
    #pragma unroll
    for (int j = 0; j < 64; j += 8) {
        float4 f0 = *(const float4*)(src + j);
        float4 f1 = *(const float4*)(src + j + 4);
        uint4 q;
        q.x = f2bf_pk(f0.x, f0.y);
        q.y = f2bf_pk(f0.z, f0.w);
        q.z = f2bf_pk(f1.x, f1.y);
        q.w = f2bf_pk(f1.z, f1.w);
        *(uint4*)(dst + j) = q;
    }
    if (s + 1 < SEQ) __builtin_prefetch(src + VD, 0, 1);  // global_prefetch_b8
}

__global__ void rnn_init_cnt(unsigned int* cnt) {
    if (threadIdx.x < LAYERS) cnt[threadIdx.x] = 0u;
}

__global__ __launch_bounds__(256)
void rnn_persistent_kernel(const float* __restrict__ x,
                           const float* __restrict__ Wx,
                           const float* __restrict__ bx,
                           const float* __restrict__ Wh,
                           const float* __restrict__ bh,
                           float* __restrict__ out_ys,       // [N][S][D]
                           float* __restrict__ out_hfin,     // [L][N][D]
                           unsigned short* __restrict__ g_h, // [L][2][N][D] bf16 staging
                           unsigned int* __restrict__ cnt)   // [L] barrier counters
{
    const int tid  = threadIdx.x;
    const int lane = tid & 31;
    const int wv   = tid >> 5;          // wave id 0..7 (wave32)
    const int l    = blockIdx.x / GPL;  // layer
    const int e0   = (blockIdx.x % GPL) * ECOLS;

    // Tile ownership: wave -> (ntile over batch rows, etile over 16-col groups)
    const int ntile = wv & 1;
    const int etile = wv >> 1;

    // LDS carve: Wx slice | Wh slice | x_t tile | h (full) — 192 KB of 320 KB
    extern __shared__ unsigned char smem_raw[];
    unsigned short* Wx_s = (unsigned short*)smem_raw;      // [ECOLS][VD]
    unsigned short* Wh_s = Wx_s + ECOLS * VD;              // [ECOLS][DD]
    unsigned short* x_s  = Wh_s + ECOLS * DD;              // [NB][VD]
    unsigned short* h_s  = x_s + NB * VD;                  // [NB][DD]

    // ---- one-time: stage weight slices (f32 -> bf16) and zero h ----
    for (int i = tid; i < ECOLS * VD; i += 256) {
        int er = i >> 9, v = i & 511;
        Wx_s[i] = f2bf(Wx[(size_t)l * DD * VD + (size_t)(e0 + er) * VD + v]);
        Wh_s[i] = f2bf(Wh[(size_t)l * DD * DD + (size_t)(e0 + er) * DD + v]);
    }
    for (int i = tid; i < NB * DD; i += 256) h_s[i] = 0;

    // Per-lane bias (bx+bh) for this lane's output column
    const int col = lane & 15;
    const int eg  = e0 + etile * 16 + col;
    const float bias = bx[l * DD + eg] + bh[l * DD + eg];

    const int arow = (lane & 15) + ntile * 16;   // A-fragment row (batch index)
    const int brow = etile * 16 + col;           // B-fragment row in W slice

    stage_x(x, x_s, 0, tid);
    __syncthreads();

    for (int s = 0; s < SEQ; ++s) {
        // ---- fused GEMMs, software-pipelined, two independent acc chains ----
        v8f accX, accH;
        #pragma unroll
        for (int j = 0; j < 8; ++j) { accX[j] = 0.0f; accH[j] = 0.0f; }

        v16bf ax  = load_frag(x_s,  arow, 0, lane);
        v16bf bxw = load_frag(Wx_s, brow, 0, lane);
        v16bf ah  = load_frag(h_s,  arow, 0, lane);
        v16bf bhw = load_frag(Wh_s, brow, 0, lane);
        #pragma unroll
        for (int kt = 0; kt < 16; ++kt) {
            v16bf axn, bxn, ahn, bhn;
            if (kt < 15) {                       // issue next frags before WMMAs
                axn = load_frag(x_s,  arow, kt + 1, lane);
                bxn = load_frag(Wx_s, brow, kt + 1, lane);
                ahn = load_frag(h_s,  arow, kt + 1, lane);
                bhn = load_frag(Wh_s, brow, kt + 1, lane);
            }
            accX = wmma_bf16(ax, bxw, accX);
            accH = wmma_bf16(ah, bhw, accH);
            if (kt < 15) { ax = axn; bxw = bxn; ah = ahn; bhw = bhn; }
        }

        // ---- sigmoid + stores (C-layout: VGPR j -> row M=j / 8+j, col = lane%16) ----
        const int buf = s & 1;
        unsigned short* gdst = g_h + ((size_t)l * 2 + buf) * NB * DD;
        #pragma unroll
        for (int j = 0; j < 8; ++j) {
            float v = 1.0f / (1.0f + __expf(-(accX[j] + accH[j] + bias)));
            int m = ((lane < 16) ? j : 8 + j) + ntile * 16;   // batch row n
            gdst[m * DD + eg] = f2bf(v);
            if (l == LAYERS - 1) out_ys[((size_t)m * SEQ + s) * DD + eg] = v;
            if (s == SEQ - 1)    out_hfin[((size_t)l * NB + m) * DD + eg] = v;
        }

        // ---- per-layer cross-WG barrier; overlap next-x staging with the spin ----
        __threadfence();
        __syncthreads();                         // all slice stores issued; x_s free
        if (tid == 0)
            __hip_atomic_fetch_add(&cnt[l], 1u, __ATOMIC_RELEASE, __HIP_MEMORY_SCOPE_AGENT);
        if (s + 1 < SEQ)
            stage_x(x, x_s, s + 1, tid);         // hide x HBM/L2 latency behind barrier
        if (tid == 0) {
            const unsigned int target = (unsigned int)GPL * (unsigned int)(s + 1);
            while (__hip_atomic_load(&cnt[l], __ATOMIC_ACQUIRE, __HIP_MEMORY_SCOPE_AGENT) < target)
                __builtin_amdgcn_s_sleep(8);
        }
        __syncthreads();

        // ---- reload full h (all 512 cols) from staging into LDS ----
        const unsigned short* gsrc = g_h + ((size_t)l * 2 + buf) * NB * DD;
        for (int i = tid * 8; i < NB * DD; i += 256 * 8)
            *(uint4*)(h_s + i) = *(const uint4*)(gsrc + i);
        __syncthreads();
    }
}

extern "C" void kernel_launch(void* const* d_in, const int* in_sizes, int n_in,
                              void* d_out, int out_size, void* d_ws, size_t ws_size,
                              hipStream_t stream) {
    const float* x  = (const float*)d_in[0];
    const float* Wx = (const float*)d_in[1];
    const float* bx = (const float*)d_in[2];
    const float* Wh = (const float*)d_in[3];
    const float* bh = (const float*)d_in[4];

    float* out_ys   = (float*)d_out;                       // [N][S][D]
    float* out_hfin = out_ys + (size_t)NB * SEQ * DD;      // [L][N][D]

    // Workspace carve: [0,256) barrier counters; then bf16 h staging (256 KB)
    unsigned int*   cnt = (unsigned int*)d_ws;
    unsigned short* g_h = (unsigned short*)((char*)d_ws + 256);

    rnn_init_cnt<<<1, 64, 0, stream>>>(cnt);

    const size_t smem = (size_t)(ECOLS * VD + ECOLS * DD + NB * VD + NB * DD)
                        * sizeof(unsigned short);          // 192 KB
    rnn_persistent_kernel<<<dim3(LAYERS * GPL), dim3(256), smem, stream>>>(
        x, Wx, bx, Wh, bh, out_ys, out_hfin, g_h, cnt);
}
